// MDA_4183298146862
// MI455X (gfx1250) — matrix operand order repaired and checked by reading, not compiled, not run on hardware
//
#include <hip/hip_runtime.h>
#include <hip/hip_bf16.h>

typedef __attribute__((ext_vector_type(16))) _Float16 v16h;
typedef __attribute__((ext_vector_type(8)))  float    v8f;

#define HW 4096
#define GB 32   // group-batches (2*16)

__device__ __forceinline__ float sigmoidf_(float t) { return 1.0f / (1.0f + __expf(-t)); }
__device__ __forceinline__ int clampi_(int v, int lo, int hi) { return v < lo ? lo : (v > hi ? hi : v); }

// ---------------------------------------------------------------------------
// Pack conv weights (O,C,3,3) f32 -> f16 in per-lane WMMA B-fragment order:
// dst[(((tap*2+kc)*NT + nt)*32 + lane)*16 + e] = w[o][c][tap]
//   o = nt*16 + (lane&15),  c = kc*32 + (e&8)*2 + (lane>>4)*8 + (e&7)
// so each B fragment is one contiguous, fully coalesced 32B load per lane.
// ---------------------------------------------------------------------------
__global__ void pack_b_frags(const float* __restrict__ w, _Float16* __restrict__ dst,
                             int NT, int Ovalid) {
    int idx = blockIdx.x * 256 + threadIdx.x;
    int total = 9 * 2 * NT * 512;
    if (idx >= total) return;
    int e   = idx & 15;
    int t   = (idx >> 4) & 31;
    int nt  = (idx >> 9) % NT;
    int kc  = (idx / (512 * NT)) & 1;
    int tap = idx / (1024 * NT);
    int o = nt * 16 + (t & 15);
    int c = kc * 32 + ((e & 8) * 2) + ((t >> 4) * 8) + (e & 7);
    float v = (o < Ovalid) ? w[(o * 64 + c) * 9 + tap] : 0.0f;
    dst[idx] = (_Float16)v;
}

// ---------------------------------------------------------------------------
// Stage 1 (fused pointwise): r=relu(BN(IR@g)); wm=IS@r; red=RW@(wm*g+eps);
// x_h[b,ch,y] = mean_x red.   One block per (b,y), 64 threads (one per x).
// ---------------------------------------------------------------------------
__global__ void stage1_kernel(const float* __restrict__ x,
                              const float* __restrict__ irw, const float* __restrict__ irb,
                              const float* __restrict__ bng, const float* __restrict__ bnb,
                              const float* __restrict__ isw, const float* __restrict__ isb,
                              const float* __restrict__ rw,  const float* __restrict__ rb,
                              float* __restrict__ xh) {
    __shared__ float s_ir[16 * 64];
    __shared__ float s_is[4 * 16];
    __shared__ float s_rw[32 * 64];
    __shared__ float s_red[64 * 32];
    int tid = threadIdx.x;                 // 64 threads
    for (int i = tid; i < 1024; i += 64) s_ir[i] = irw[i];
    if (tid < 64) s_is[tid] = isw[tid];
    for (int i = tid; i < 2048; i += 64) s_rw[i] = rw[i];
    int b = blockIdx.x >> 6, y = blockIdx.x & 63;
    __syncthreads();

    const float* xb = x + (((size_t)b * 64) * 64 + y) * 64 + tid;
    float g[64];
#pragma unroll
    for (int c = 0; c < 64; ++c) g[c] = xb[(size_t)c * HW];

    float r[16];
#pragma unroll
    for (int o = 0; o < 16; ++o) {
        float a = irb[o];
#pragma unroll
        for (int c = 0; c < 64; ++c) a += s_ir[o * 64 + c] * g[c];
        a = a * (bng[o] * rsqrtf(1.0f + 1e-5f)) + bnb[o];
        r[o] = fmaxf(a, 0.0f);
    }
    float wm[4];
#pragma unroll
    for (int o = 0; o < 4; ++o) {
        float a = isb[o];
#pragma unroll
        for (int c = 0; c < 16; ++c) a += s_is[o * 16 + c] * r[c];
        wm[o] = a;
    }
#pragma unroll 4
    for (int oc = 0; oc < 32; ++oc) {
        float a = rb[oc];
#pragma unroll
        for (int c = 0; c < 64; ++c)
            a += s_rw[oc * 64 + c] * (wm[c >> 4] * g[c] + 1.1920929e-07f);
        s_red[tid * 32 + oc] = a;
    }
    __syncthreads();
    if (tid < 32) {
        float s = 0.0f;
        for (int p = 0; p < 64; ++p) s += s_red[p * 32 + tid];
        xh[((size_t)b * 32 + tid) * 64 + y] = s * (1.0f / 64.0f);
    }
}

// ---------------------------------------------------------------------------
// Row gate: f[b,o,y] = sigmoid(sigmoid(res_w @ x_h[b,:,y] + res_b))
// (the x_w half of the reference concat is dead code: hw[:,:,:h] only reads
// positions produced from x_h). One block per (b,y), 64 threads.
// ---------------------------------------------------------------------------
__global__ void rowgate_kernel(const float* __restrict__ xh,
                               const float* __restrict__ resw, const float* __restrict__ resb,
                               float* __restrict__ f) {
    __shared__ float v[32];
    int b = blockIdx.x >> 6, y = blockIdx.x & 63, o = threadIdx.x;
    if (o < 32) v[o] = xh[((size_t)b * 32 + o) * 64 + y];
    __syncthreads();
    float a = resb[o];
#pragma unroll
    for (int c = 0; c < 32; ++c) a += resw[o * 32 + c] * v[c];
    f[((size_t)b * 64 + o) * 64 + y] = sigmoidf_(sigmoidf_(a));
}

// ---------------------------------------------------------------------------
// Channel attention reductions: out = gx*f ;  avg/max over HxW per (b,c).
// ---------------------------------------------------------------------------
__global__ void ca_reduce_kernel(const float* __restrict__ x, const float* __restrict__ f,
                                 float* __restrict__ avg, float* __restrict__ mx) {
    __shared__ float ss[256], sm[256];
    int bc = blockIdx.x, tid = threadIdx.x;
    const float* xp = x + (size_t)bc * HW;
    const float* fp = f + (size_t)bc * 64;
    float s = 0.0f, m = -3.4e38f;
    for (int i = tid; i < HW; i += 256) {
        float v = xp[i] * fp[i >> 6];
        s += v; m = fmaxf(m, v);
    }
    ss[tid] = s; sm[tid] = m; __syncthreads();
    for (int st = 128; st > 0; st >>= 1) {
        if (tid < st) { ss[tid] += ss[tid + st]; sm[tid] = fmaxf(sm[tid], sm[tid + st]); }
        __syncthreads();
    }
    if (tid == 0) { avg[bc] = ss[0] * (1.0f / HW); mx[bc] = sm[0]; }
}

// ca[b,o] = sigmoid( fc2@relu(fc1@avg) + fc2@relu(fc1@mx) )
__global__ void ca_fc_kernel(const float* __restrict__ avg, const float* __restrict__ mx,
                             const float* __restrict__ fc1, const float* __restrict__ fc2,
                             float* __restrict__ ca) {
    __shared__ float sa[64], sm[64], h[16];
    int b = blockIdx.x, tid = threadIdx.x;     // 64 threads
    sa[tid] = avg[b * 64 + tid]; sm[tid] = mx[b * 64 + tid];
    __syncthreads();
    if (tid < 16) {
        float a = 0.0f, m = 0.0f;
#pragma unroll
        for (int c = 0; c < 64; ++c) { a += fc1[tid * 64 + c] * sa[c]; m += fc1[tid * 64 + c] * sm[c]; }
        h[tid] = fmaxf(a, 0.0f) + fmaxf(m, 0.0f);
    }
    __syncthreads();
    float s = 0.0f;
#pragma unroll
    for (int c = 0; c < 16; ++c) s += fc2[tid * 16 + c] * h[c];
    ca[b * 64 + tid] = sigmoidf_(s);
}

// ---------------------------------------------------------------------------
// 3x3 conv as implicit GEMM on v_wmma_f32_16x16x32_f16.
// Block = one image row (b,y): NT waves, wave w owns out-channels [16w,16w+16).
// LDS A-tile: 3 rows x 66 cols x 64ch in f16, K-contiguous per pixel.
// K loop flattened to 18 steps (9 taps x 2 K-halves), software-pipelined two
// deep with distinct register buffers so ds_load/global_load of step i+1
// overlap the 4 WMMAs of step i (no s_wait_dscnt 0 serialization).
// ---------------------------------------------------------------------------
__global__ void conv3x3_wmma_kernel(const float* __restrict__ in, const _Float16* __restrict__ Wp,
                                    const float* __restrict__ bias, float* __restrict__ out,
                                    int NT, int Ovalid) {
    extern __shared__ char smem_raw[];
    _Float16* Ald = (_Float16*)smem_raw;           // [3][66][64]
    int tid = threadIdx.x, nthr = blockDim.x;
    int b = blockIdx.x >> 6, y = blockIdx.x & 63;

    for (int i = tid; i < 3 * 66 * 64; i += nthr) {
        int col = i % 66;
        int ch  = (i / 66) & 63;
        int row = i / (66 * 64);
        int gy = y + row - 1, gxx = col - 1;
        float v = 0.0f;
        if ((unsigned)gy < 64u && (unsigned)gxx < 64u)
            v = in[((size_t)(b * 64 + ch) * 64 + gy) * 64 + gxx];
        Ald[(row * 66 + col) * 64 + ch] = (_Float16)v;
    }
    __syncthreads();

    int wid = tid >> 5, lane = tid & 31, khalf = lane >> 4;

    auto loadB = [&](int step, v16h& bf) {
        const _Float16* bp = Wp + ((size_t)((step * NT + wid) * 32 + lane) << 4);
        ((uint4*)&bf)[0] = ((const uint4*)bp)[0];
        ((uint4*)&bf)[1] = ((const uint4*)bp)[1];
    };
    auto loadA = [&](int step, v16h* af) {
        int tap = step >> 1, kc = step & 1;
        int ky = tap / 3, kx = tap % 3;
        const _Float16* base = Ald + (ky * 66 + (lane & 15) + kx) * 64 + kc * 32 + khalf * 8;
#pragma unroll
        for (int mt = 0; mt < 4; ++mt) {
            const _Float16* ap = base + mt * 16 * 64;   // col += 16 per M tile
            ((uint4*)&af[mt])[0] = *(const uint4*)ap;
            ((uint4*)&af[mt])[1] = *(const uint4*)(ap + 16);
        }
    };

    v8f acc[4];
#pragma unroll
    for (int mt = 0; mt < 4; ++mt)
#pragma unroll
        for (int j = 0; j < 8; ++j) acc[mt][j] = 0.0f;

    v16h bf[2];
    v16h af[2][4];
    loadB(0, bf[0]);
    loadA(0, af[0]);
#pragma unroll
    for (int i = 0; i < 18; ++i) {
        int cur = i & 1, nxt = cur ^ 1;
        if (i < 17) { loadB(i + 1, bf[nxt]); loadA(i + 1, af[nxt]); }
#pragma unroll
        for (int mt = 0; mt < 4; ++mt)
            acc[mt] = __builtin_amdgcn_wmma_f32_16x16x32_f16(
                false, af[cur][mt], false, bf[cur], (short)0, acc[mt], false, false);
    }

    int o = wid * 16 + (lane & 15);                 // D: N = lane&15
    float bv = (o < Ovalid) ? bias[o] : 0.0f;
    int OC = NT * 16;
#pragma unroll
    for (int mt = 0; mt < 4; ++mt) {
        float* op = out + ((size_t)(b * OC + o) * 64 + y) * 64 + mt * 16 + khalf * 8;
#pragma unroll
        for (int j = 0; j < 8; ++j) op[j] = acc[mt][j] + bv;   // D: M = j + khalf*8
    }
}

// ---------------------------------------------------------------------------
// Instance-norm stats + apply (per (b,c) over HxW), biased variance.
// ---------------------------------------------------------------------------
__global__ void gn_stats_kernel(const float* __restrict__ xr2, float* __restrict__ mu,
                                float* __restrict__ rs) {
    __shared__ float ss[256], sq[256];
    int bc = blockIdx.x, tid = threadIdx.x;
    const float* p = xr2 + (size_t)bc * HW;
    float s = 0.0f, q = 0.0f;
    for (int i = tid; i < HW; i += 256) { float v = p[i]; s += v; q += v * v; }
    ss[tid] = s; sq[tid] = q; __syncthreads();
    for (int st = 128; st > 0; st >>= 1) {
        if (tid < st) { ss[tid] += ss[tid + st]; sq[tid] += sq[tid + st]; }
        __syncthreads();
    }
    if (tid == 0) {
        float m = ss[0] * (1.0f / HW);
        float v = sq[0] * (1.0f / HW) - m * m;
        mu[bc] = m; rs[bc] = rsqrtf(v + 1e-5f);
    }
}

__global__ void gn_apply_kernel(float* __restrict__ xr2, const float* __restrict__ mu,
                                const float* __restrict__ rs, const float* __restrict__ gg,
                                const float* __restrict__ gb) {
    size_t i = (size_t)blockIdx.x * 256 + threadIdx.x;
    if (i >= (size_t)GB * 64 * HW) return;
    int bc = (int)(i >> 12);
    int c  = bc & 63;
    xr2[i] = (xr2[i] - mu[bc]) * rs[bc] * gg[c] + gb[c];
}

// ---------------------------------------------------------------------------
// Deformable conv (WMMA) + fused final epilogue.
// Block per (b,y), 4 waves. Per 16-pixel M tile: sample 16x9x64 bilinear taps
// (mask-scaled, f16) into LDS as A[pixel][K=tap*64+c], then a software-
// pipelined chain of 18 WMMAs/wave (K=576), then fused:
//   out = gx * sigmoid(gx*f + ca*relu(dcn+b)).
// ---------------------------------------------------------------------------
__global__ void dcn_wmma_kernel(const float* __restrict__ xn, const float* __restrict__ off,
                                const _Float16* __restrict__ Wp, const float* __restrict__ dcnb,
                                const float* __restrict__ ca, const float* __restrict__ x,
                                const float* __restrict__ f, float* __restrict__ outp) {
    __shared__ __align__(16) _Float16 As[16 * 9 * 64];
    int tid = threadIdx.x;                 // 128 threads, 4 waves
    int b = blockIdx.x >> 6, y = blockIdx.x & 63;
    int wid = tid >> 5, lane = tid & 31, khalf = lane >> 4;

    auto loadB = [&](int step, v16h& bf) {
        const _Float16* bp = Wp + ((size_t)((step * 4 + wid) * 32 + lane) << 4);
        ((uint4*)&bf)[0] = ((const uint4*)bp)[0];
        ((uint4*)&bf)[1] = ((const uint4*)bp)[1];
    };
    auto loadA = [&](int step, v16h& afr) {
        int tap = step >> 1, kc = step & 1;
        const _Float16* ap = As + ((lane & 15) * 9 + tap) * 64 + kc * 32 + khalf * 8;
        ((uint4*)&afr)[0] = *(const uint4*)ap;
        ((uint4*)&afr)[1] = *(const uint4*)(ap + 16);
    };

    for (int mt = 0; mt < 4; ++mt) {
        // ---- phase A: bilinear sample into LDS A tile ----
        for (int pi = tid; pi < 144; pi += 128) {
            int p = pi / 9, k = pi % 9;
            int xcol = mt * 16 + p;
            int ky = k / 3 - 1, kx = k % 3 - 1;
            size_t obase = (((size_t)b * 32) * 64 + y) * 64 + xcol;
            float dy = off[obase + (size_t)(2 * k) * HW];
            float dx = off[obase + (size_t)(2 * k + 1) * HW];
            float mv = sigmoidf_(off[obase + (size_t)(18 + k) * HW]);
            float py = (float)(y + ky) + dy;
            float px = (float)(xcol + kx) + dx;
            float y0 = floorf(py), x0 = floorf(px);
            float wy = py - y0, wx = px - x0;
            int y0i = (int)y0, x0i = (int)x0;
            int yc0 = clampi_(y0i, 0, 63), yc1 = clampi_(y0i + 1, 0, 63);
            int xc0 = clampi_(x0i, 0, 63), xc1 = clampi_(x0i + 1, 0, 63);
            float v00 = ((unsigned)y0i < 64u && (unsigned)x0i < 64u) ? 1.0f : 0.0f;
            float v01 = ((unsigned)y0i < 64u && (unsigned)(x0i + 1) < 64u) ? 1.0f : 0.0f;
            float v10 = ((unsigned)(y0i + 1) < 64u && (unsigned)x0i < 64u) ? 1.0f : 0.0f;
            float v11 = ((unsigned)(y0i + 1) < 64u && (unsigned)(x0i + 1) < 64u) ? 1.0f : 0.0f;
            float w00 = (1.0f - wy) * (1.0f - wx) * v00;
            float w01 = (1.0f - wy) * wx * v01;
            float w10 = wy * (1.0f - wx) * v10;
            float w11 = wy * wx * v11;
            int i00 = yc0 * 64 + xc0, i01 = yc0 * 64 + xc1;
            int i10 = yc1 * 64 + xc0, i11 = yc1 * 64 + xc1;
            const float* xc = xn + ((size_t)b * 64) * HW;
            _Float16* dst = As + (p * 9 + k) * 64;
            for (int c = 0; c < 64; ++c) {
                const float* cp = xc + (size_t)c * HW;
                float v = w00 * cp[i00] + w01 * cp[i01] + w10 * cp[i10] + w11 * cp[i11];
                dst[c] = (_Float16)(v * mv);
            }
        }
        __syncthreads();

        // ---- phase B: pipelined 18-step WMMA chain ----
        v8f acc;
#pragma unroll
        for (int j = 0; j < 8; ++j) acc[j] = 0.0f;
        v16h bf[2], afr[2];
        loadB(0, bf[0]);
        loadA(0, afr[0]);
#pragma unroll
        for (int i = 0; i < 18; ++i) {
            int cur = i & 1, nxt = cur ^ 1;
            if (i < 17) { loadB(i + 1, bf[nxt]); loadA(i + 1, afr[nxt]); }
            acc = __builtin_amdgcn_wmma_f32_16x16x32_f16(
                false, afr[cur], false, bf[cur], (short)0, acc, false, false);
        }

        // ---- fused epilogue: final = gx * sigmoid(gx*f + ca*relu(dcn)) ----
        int o = wid * 16 + (lane & 15);
        float bv = dcnb[o];
        float cav = ca[b * 64 + o];
        float fv = f[((size_t)b * 64 + o) * 64 + y];
        size_t gbase = ((size_t)(b * 64 + o) * 64 + y) * 64 + mt * 16 + khalf * 8;
#pragma unroll
        for (int j = 0; j < 8; ++j) {
            float d = fmaxf(acc[j] + bv, 0.0f) * cav;
            float g = x[gbase + j];
            float t = g * fv + d;
            outp[gbase + j] = g * sigmoidf_(t);
        }
        __syncthreads();
    }
}

// ---------------------------------------------------------------------------
extern "C" void kernel_launch(void* const* d_in, const int* in_sizes, int n_in,
                              void* d_out, int out_size, void* d_ws, size_t ws_size,
                              hipStream_t stream) {
    (void)in_sizes; (void)n_in; (void)out_size; (void)ws_size;
    const float* x     = (const float*)d_in[0];
    const float* irw   = (const float*)d_in[1];
    const float* irb   = (const float*)d_in[2];
    const float* bng   = (const float*)d_in[3];
    const float* bnb   = (const float*)d_in[4];
    const float* isw   = (const float*)d_in[5];
    const float* isb   = (const float*)d_in[6];
    const float* rw    = (const float*)d_in[7];
    const float* rb    = (const float*)d_in[8];
    const float* resw  = (const float*)d_in[9];
    const float* resb  = (const float*)d_in[10];
    const float* fc1   = (const float*)d_in[11];
    const float* fc2   = (const float*)d_in[12];
    const float* c3w   = (const float*)d_in[13];
    const float* c3b   = (const float*)d_in[14];
    const float* gng   = (const float*)d_in[15];
    const float* gnb   = (const float*)d_in[16];
    const float* offw  = (const float*)d_in[17];
    const float* offb  = (const float*)d_in[18];
    const float* dcnw  = (const float*)d_in[19];
    const float* dcnb  = (const float*)d_in[20];
    float* out = (float*)d_out;

    char* ws = (char*)d_ws;
    size_t off = 0;
    auto alloc = [&](size_t bytes) -> char* {
        char* p = ws + off;
        off = (off + bytes + 255) & ~(size_t)255;
        return p;
    };
    _Float16* WpC3  = (_Float16*)alloc(9 * 2 * 4 * 512 * sizeof(_Float16)); // 72KB
    _Float16* WpOff = (_Float16*)alloc(9 * 2 * 2 * 512 * sizeof(_Float16)); // 36KB
    _Float16* WpDcn = (_Float16*)alloc(9 * 2 * 4 * 512 * sizeof(_Float16)); // 72KB
    float* xh   = (float*)alloc((size_t)GB * 32 * 64 * 4);
    float* fbuf = (float*)alloc((size_t)GB * 64 * 64 * 4);
    float* avg  = (float*)alloc((size_t)GB * 64 * 4);
    float* mxb  = (float*)alloc((size_t)GB * 64 * 4);
    float* cab  = (float*)alloc((size_t)GB * 64 * 4);
    float* mu   = (float*)alloc((size_t)GB * 64 * 4);
    float* rs   = (float*)alloc((size_t)GB * 64 * 4);
    float* xr2  = (float*)alloc((size_t)GB * 64 * HW * 4);   // 32MB (raw then normed in place)
    float* offo = (float*)alloc((size_t)GB * 32 * HW * 4);   // 16MB (27 used, padded to 32)

    // 1) pack conv weights into WMMA B-fragment order (f16)
    pack_b_frags<<<144, 256, 0, stream>>>(c3w,  WpC3,  4, 64);
    pack_b_frags<<<72,  256, 0, stream>>>(offw, WpOff, 2, 27);
    pack_b_frags<<<144, 256, 0, stream>>>(dcnw, WpDcn, 4, 64);

    // 2) fused pointwise stage -> x_h row means
    stage1_kernel<<<GB * 64, 64, 0, stream>>>(x, irw, irb, bng, bnb, isw, isb, rw, rb, xh);

    // 3) row gate f = sigmoid(sigmoid(res_w @ x_h))
    rowgate_kernel<<<GB * 64, 64, 0, stream>>>(xh, resw, resb, fbuf);

    // 4-5) channel attention
    ca_reduce_kernel<<<GB * 64, 256, 0, stream>>>(x, fbuf, avg, mxb);
    ca_fc_kernel<<<GB, 64, 0, stream>>>(avg, mxb, fc1, fc2, cab);

    // 6) x_r2 = conv3x3(gx, c3_w)+b   [WMMA]
    conv3x3_wmma_kernel<<<GB * 64, 128, 3 * 66 * 64 * sizeof(_Float16), stream>>>(
        x, WpC3, c3b, xr2, 4, 64);

    // 7-8) instance norm
    gn_stats_kernel<<<GB * 64, 256, 0, stream>>>(xr2, mu, rs);
    gn_apply_kernel<<<(GB * 64 * HW) / 256, 256, 0, stream>>>(xr2, mu, rs, gng, gnb);

    // 9) offsets/mask = conv3x3(x_r2, off_w)+b   [WMMA, O padded 27->32]
    conv3x3_wmma_kernel<<<GB * 64, 64, 3 * 66 * 64 * sizeof(_Float16), stream>>>(
        xr2, WpOff, offb, offo, 2, 27);

    // 10) deformable conv [WMMA] + fused final gating -> d_out
    dcn_wmma_kernel<<<GB * 64, 128, 0, stream>>>(
        xr2, offo, WpDcn, dcnb, cab, x, fbuf, out);
}
